// TransformerEncoder_2465311228124
// MI455X (gfx1250) — matrix operand order repaired and checked by reading, not compiled, not run on hardware
//
#include <hip/hip_runtime.h>
#include <hip/hip_bf16.h>
#include <cstdint>

// ---------------------------------------------------------------------------
// Types for CDNA5 WMMA (wave32): v_wmma_f32_16x16x32_bf16
// ---------------------------------------------------------------------------
typedef __bf16 bf16;
typedef __bf16 v16bf __attribute__((ext_vector_type(16)));
typedef __bf16 v8bf  __attribute__((ext_vector_type(8)));
typedef float  v8f   __attribute__((ext_vector_type(8)));

// Model constants
constexpr int kT   = 32;    // tokens per instance (NT)
constexpr int kD   = 256;   // model dim (2*EMB)
constexpr int kE   = 128;   // EMB
constexpr int kDH  = 64;    // head dim
constexpr int kB   = 8;
constexpr int kS   = 512;
constexpr int kInst = kB * kS;              // 4096
constexpr int kHistElems = kInst * kT;      // 131072

// ---------------------------------------------------------------------------
// One 16x16 f32 accumulator tile: D += A(16xK, bf16, row-major lda)
//                                  * B^T where B rows are W[n][k] (row-major ldb)
// A lane layout (16-bit A 16x32): m = lane&15; lanes<16 cover K {0..7,16..23},
// lanes>=16 cover K {8..15,24..31} per 32-wide K chunk -> two 16B loads.
// B lane layout (32x16): n = lane&15; lanes<16 cover K {0..7,16..23},
// lanes>=16 cover K {8..15,24..31} per 32-wide K chunk -> two 16B loads.
// ---------------------------------------------------------------------------
__device__ inline v8f wmma_acc_16x16(const bf16* A, int lda,
                                     const bf16* B, int ldb,
                                     int K, v8f acc) {
  const int lane = threadIdx.x & 31;
  const int am = lane & 15;
  const int ak = (lane >> 4) * 8;
  const int bn = lane & 15;
  const int bk = (lane >> 4) * 8;
  for (int k0 = 0; k0 < K; k0 += 32) {
    const bf16* ap = A + am * lda + k0 + ak;
    v8bf a_lo = *(const v8bf*)(ap);
    v8bf a_hi = *(const v8bf*)(ap + 16);
    v16bf av;
#pragma unroll
    for (int i = 0; i < 8; ++i) { av[i] = a_lo[i]; av[i + 8] = a_hi[i]; }
    const bf16* bp = B + bn * ldb + k0 + bk;
    v8bf b_lo = *(const v8bf*)(bp);
    v8bf b_hi = *(const v8bf*)(bp + 16);
    v16bf bv;
#pragma unroll
    for (int i = 0; i < 8; ++i) { bv[i] = b_lo[i]; bv[i + 8] = b_hi[i]; }
    acc = __builtin_amdgcn_wmma_f32_16x16x32_bf16(
        /*neg_a=*/false, av, /*neg_b=*/false, bv,
        /*c_mod=*/(short)0, acc, /*reuse_a=*/false, /*reuse_b=*/false);
  }
  return acc;
}

// C/D f32 layout: VGPR r -> M = r + 8*(lane>=16); N = lane&15
__device__ inline void store_tile_f32(float* out, int ldo, int m0, int n0, v8f acc) {
  const int lane = threadIdx.x & 31;
  const int n = lane & 15;
  const int mb = (lane >> 4) * 8;
#pragma unroll
  for (int r = 0; r < 8; ++r) out[(m0 + mb + r) * ldo + n0 + n] = acc[r];
}

// ---------------------------------------------------------------------------
// Stage 0: convert all weights f32 -> bf16 packed contiguously in ws
//   [in_proj_w 2*768*256][out_proj_w 2*256*256][ff1_w 2*256*256][ff2_w 2*256*256]
// ---------------------------------------------------------------------------
__global__ __launch_bounds__(256) void convert_kernel(
    const float* __restrict__ wqkv, const float* __restrict__ wo,
    const float* __restrict__ wf1, const float* __restrict__ wf2,
    bf16* __restrict__ out) {
  int i = blockIdx.x * 256 + threadIdx.x;
  float v;
  if (i < 393216) v = wqkv[i];
  else if (i < 524288) v = wo[i - 393216];
  else if (i < 655360) v = wf1[i - 524288];
  else v = wf2[i - 655360];
  out[i] = (bf16)v;
}

// ---------------------------------------------------------------------------
// Stage 1: temporal encoding te[inst][e] (npm folded in), one-hot -> idx,
// and npm output. One block per instance, 128 threads = e.
// ---------------------------------------------------------------------------
__global__ __launch_bounds__(128) void prep_kernel(
    const float* __restrict__ event_type, const float* __restrict__ event_time,
    float* __restrict__ te, int* __restrict__ idx, float* __restrict__ npm_out) {
  const int inst = blockIdx.x;
  const int e = threadIdx.x;
  __shared__ int sidx;
  if (e == 0) sidx = 0;
  __syncthreads();
  if (e < kT) {
    if (event_type[(size_t)inst * kT + e] > 0.5f) sidx = e;
  }
  __syncthreads();
  const float t = event_time[inst];
  const float npm = (t != 0.0f) ? 1.0f : 0.0f;
  if (e == 0) {
    idx[inst] = sidx;
    npm_out[inst] = npm;
  }
  const float pv = __powf(10000.0f, 2.0f * (float)(e >> 1) / (float)kE);
  const float ang = t / pv;
  const float v = ((e & 1) == 0) ? __sinf(ang) : __cosf(ang);
  te[(size_t)inst * kE + e] = v * npm;
}

// ---------------------------------------------------------------------------
// Stage 2: causal type-gated cumsum over s.
// 32768 independent chains (b,t,e); one block per (b,t), thread = e.
// ---------------------------------------------------------------------------
__global__ __launch_bounds__(128) void scan_kernel(
    const float* __restrict__ te, const int* __restrict__ idx,
    float* __restrict__ xte) {
  const int b = blockIdx.x >> 5;   // 32 t-chains per b
  const int t = blockIdx.x & 31;
  const int e = threadIdx.x;
  float acc = 0.0f;
  for (int s = 0; s < kS; ++s) {
    const int inst = b * kS + s;
    const float v = te[(size_t)inst * kE + e];
    acc += (idx[inst] == t) ? v : 0.0f;
    xte[((size_t)inst * kT + t) * kE + e] = acc;
  }
}

// ---------------------------------------------------------------------------
// Stage 3: full 2-layer transformer per instance, entirely in LDS.
// One 128-thread block (4 waves) per instance; WMMA bf16 for every GEMM.
// LDS plan (144 KB total -> 2 blocks per 320 KB WGP = 8 waves/WGP):
//   [     0, 32768) xf   f32 [32][256]  activations (residual), persistent
//   [ 32768, 49152) xb   bf16[32][256]  activations bf16 (WMMA A), persistent
//   [ 49152, 65536) qb   bf16[32][256]  q (dead after scores) -> aliased as ob
//   [ 65536, 81920) kb   bf16[32][256]  k (dead after scores) -> aliased as ab
//   [ 81920, 98304) vt   bf16[256][32]  v transposed [dim][token]
//   [ 98304,114688) sf   f32 [4][32][32] scores
//   [114688,147456) of   f32 [32][256]  GEMM f32 output / LN temp
// ---------------------------------------------------------------------------
__global__ __launch_bounds__(128) void tx_kernel(
    const float* __restrict__ cat_emb, const float* __restrict__ xte,
    const bf16* __restrict__ wqkv, const float* __restrict__ bqkv,
    const bf16* __restrict__ wo, const float* __restrict__ bo,
    const bf16* __restrict__ wf1, const float* __restrict__ bf1,
    const bf16* __restrict__ wf2, const float* __restrict__ bf2,
    const float* __restrict__ ln1w, const float* __restrict__ ln1b,
    const float* __restrict__ ln2w, const float* __restrict__ ln2b,
    const float* __restrict__ histw, const float* __restrict__ histb,
    float* __restrict__ out_hist) {
  __shared__ __align__(16) char smem[147456];
  float* xf = (float*)(smem);
  bf16*  xb = (bf16*)(smem + 32768);
  bf16*  qb = (bf16*)(smem + 49152);
  bf16*  kb = (bf16*)(smem + 65536);
  bf16*  vt = (bf16*)(smem + 81920);
  float* sf = (float*)(smem + 98304);
  float* of = (float*)(smem + 114688);
  bf16*  ob = qb;   // alias: ob live only after qb is dead (post-scores)
  bf16*  ab = kb;   // alias: ab live only after kb is dead (post-scores)

  const int inst = blockIdx.x;
  const int tid = threadIdx.x;
  const int wave = tid >> 5;
  const int lane = tid & 31;
  const int n_ = lane & 15;
  const int mb_ = (lane >> 4) * 8;

  // Build x = [cat_emb | x_t_emb]
  for (int i = tid; i < kT * kD; i += 128) {
    const int m = i >> 8, c = i & 255;
    float v = (c < kE) ? cat_emb[m * kE + c]
                       : xte[((size_t)inst * kT + m) * kE + (c - kE)];
    xf[i] = v;
    xb[i] = (bf16)v;
  }
  __syncthreads();

  for (int l = 0; l < 2; ++l) {
    const bf16* Wq = wqkv + (size_t)l * 768 * 256;
    const float* Bq = bqkv + l * 768;
    // ---- qkv = x @ Wq^T + b : 2 x 48 tiles, uniform routing on nt ----
    for (int t = wave; t < 2 * 48; t += 4) {
      const int mt = t & 1, nt = t >> 1;
      v8f acc = {};
      acc = wmma_acc_16x16(xb + mt * 16 * kD, kD, Wq + (size_t)(nt * 16) * kD, kD, kD, acc);
      const int ng = nt * 16 + n_;
      const float bias = Bq[ng];
      if (nt < 16) {                 // q (fold 1/sqrt(dh) = 1/8)
        const int c = ng;
#pragma unroll
        for (int r = 0; r < 8; ++r)
          qb[(mt * 16 + mb_ + r) * kD + c] = (bf16)((acc[r] + bias) * 0.125f);
      } else if (nt < 32) {          // k
        const int c = ng - 256;
#pragma unroll
        for (int r = 0; r < 8; ++r)
          kb[(mt * 16 + mb_ + r) * kD + c] = (bf16)(acc[r] + bias);
      } else {                       // v, stored transposed [dim][token]
        const int c = ng - 512;
#pragma unroll
        for (int r = 0; r < 8; ++r)
          vt[c * kT + (mt * 16 + mb_ + r)] = (bf16)(acc[r] + bias);
      }
    }
    __syncthreads();

    // ---- scores[h] = q_h @ k_h^T : wave == head, 2x2 tiles of K=64 ----
    {
      const int h = wave;
#pragma unroll
      for (int mt = 0; mt < 2; ++mt)
#pragma unroll
        for (int nt = 0; nt < 2; ++nt) {
          v8f acc = {};
          acc = wmma_acc_16x16(qb + (mt * 16) * kD + h * kDH, kD,
                               kb + (nt * 16) * kD + h * kDH, kD, kDH, acc);
#pragma unroll
          for (int r = 0; r < 8; ++r)
            sf[(h * kT + mt * 16 + mb_ + r) * kT + nt * 16 + n_] = acc[r];
        }
    }
    __syncthreads();

    // ---- softmax: one thread per (head,row), f32 -> bf16 attn ----
    {
      const int h = tid >> 5, row = tid & 31;
      float vals[kT];
      float mx = -1e30f;
#pragma unroll
      for (int j = 0; j < kT; ++j) {
        vals[j] = sf[(h * kT + row) * kT + j];
        mx = fmaxf(mx, vals[j]);
      }
      float sum = 0.0f;
#pragma unroll
      for (int j = 0; j < kT; ++j) { vals[j] = __expf(vals[j] - mx); sum += vals[j]; }
      const float inv = 1.0f / sum;
#pragma unroll
      for (int j = 0; j < kT; ++j)
        ab[(h * kT + row) * kT + j] = (bf16)(vals[j] * inv);
    }
    __syncthreads();

    // ---- o = attn @ v : 2 x 16 tiles, K = 32 (one WMMA each) ----
    for (int t = wave; t < 2 * 16; t += 4) {
      const int mt = t & 1, nt = t >> 1;
      const int h = nt >> 2;   // 64 output dims per head = 4 ntiles
      v8f acc = {};
      acc = wmma_acc_16x16(ab + (h * kT + mt * 16) * kT, kT,
                           vt + (nt * 16) * kT, kT, kT, acc);
      store_tile_f32(of, kD, mt * 16, nt * 16, acc);
    }
    __syncthreads();

    // ---- ob = bf16(of)  (ob aliases qb; qb dead) ----
    for (int i = tid; i < kT * kD; i += 128) ob[i] = (bf16)of[i];
    __syncthreads();

    // ---- out_proj -> of (f32) ----
    const bf16* Wol = wo + (size_t)l * kD * kD;
    const float* Bol = bo + l * kD;
    for (int t = wave; t < 2 * 16; t += 4) {
      const int mt = t & 1, nt = t >> 1;
      v8f acc = {};
      acc = wmma_acc_16x16(ob + mt * 16 * kD, kD, Wol + (size_t)(nt * 16) * kD, kD, kD, acc);
      const int ng = nt * 16 + n_;
      const float bias = Bol[ng];
#pragma unroll
      for (int r = 0; r < 8; ++r) of[(mt * 16 + mb_ + r) * kD + ng] = acc[r] + bias;
    }
    __syncthreads();

    // ---- x = LN1(x + o) ----
    if (tid < kT) {
      const float* w = ln1w + l * kD;
      const float* b = ln1b + l * kD;
      float s = 0.0f;
      for (int c = 0; c < kD; ++c) {
        float v = xf[tid * kD + c] + of[tid * kD + c];
        of[tid * kD + c] = v;
        s += v;
      }
      const float mean = s * (1.0f / kD);
      float var = 0.0f;
      for (int c = 0; c < kD; ++c) { float d = of[tid * kD + c] - mean; var += d * d; }
      const float rinv = rsqrtf(var * (1.0f / kD) + 1e-5f);
      for (int c = 0; c < kD; ++c) {
        const float nv = (of[tid * kD + c] - mean) * rinv * w[c] + b[c];
        xf[tid * kD + c] = nv;
        xb[tid * kD + c] = (bf16)nv;
      }
    }
    __syncthreads();

    // ---- ff1 + relu -> ob (bf16 directly) ----
    const bf16* Wf1 = wf1 + (size_t)l * kD * kD;
    const float* Bf1 = bf1 + l * kD;
    for (int t = wave; t < 2 * 16; t += 4) {
      const int mt = t & 1, nt = t >> 1;
      v8f acc = {};
      acc = wmma_acc_16x16(xb + mt * 16 * kD, kD, Wf1 + (size_t)(nt * 16) * kD, kD, kD, acc);
      const int ng = nt * 16 + n_;
      const float bias = Bf1[ng];
#pragma unroll
      for (int r = 0; r < 8; ++r)
        ob[(mt * 16 + mb_ + r) * kD + ng] = (bf16)fmaxf(acc[r] + bias, 0.0f);
    }
    __syncthreads();

    // ---- ff2 -> of (f32) ----
    const bf16* Wf2 = wf2 + (size_t)l * kD * kD;
    const float* Bf2 = bf2 + l * kD;
    for (int t = wave; t < 2 * 16; t += 4) {
      const int mt = t & 1, nt = t >> 1;
      v8f acc = {};
      acc = wmma_acc_16x16(ob + mt * 16 * kD, kD, Wf2 + (size_t)(nt * 16) * kD, kD, kD, acc);
      const int ng = nt * 16 + n_;
      const float bias = Bf2[ng];
#pragma unroll
      for (int r = 0; r < 8; ++r) of[(mt * 16 + mb_ + r) * kD + ng] = acc[r] + bias;
    }
    __syncthreads();

    // ---- x = LN2(x + ff) ----
    if (tid < kT) {
      const float* w = ln2w + l * kD;
      const float* b = ln2b + l * kD;
      float s = 0.0f;
      for (int c = 0; c < kD; ++c) {
        float v = xf[tid * kD + c] + of[tid * kD + c];
        of[tid * kD + c] = v;
        s += v;
      }
      const float mean = s * (1.0f / kD);
      float var = 0.0f;
      for (int c = 0; c < kD; ++c) { float d = of[tid * kD + c] - mean; var += d * d; }
      const float rinv = rsqrtf(var * (1.0f / kD) + 1e-5f);
      for (int c = 0; c < kD; ++c) {
        const float nv = (of[tid * kD + c] - mean) * rinv * w[c] + b[c];
        xf[tid * kD + c] = nv;
        xb[tid * kD + c] = (bf16)nv;
      }
    }
    __syncthreads();
  }

  // ---- head: sigmoid(x @ hist_w^T + b) ----
  if (tid < kT) {
    float s = histb[0];
    for (int c = 0; c < kD; ++c) s += xf[tid * kD + c] * histw[c];
    out_hist[(size_t)inst * kT + tid] = 1.0f / (1.0f + __expf(-s));
  }
}

// ---------------------------------------------------------------------------
// Host launcher
// ---------------------------------------------------------------------------
extern "C" void kernel_launch(void* const* d_in, const int* in_sizes, int n_in,
                              void* d_out, int out_size, void* d_ws, size_t ws_size,
                              hipStream_t stream) {
  (void)in_sizes; (void)n_in; (void)out_size; (void)ws_size;
  const float* event_type = (const float*)d_in[0];
  const float* event_time = (const float*)d_in[1];
  const float* cat_emb    = (const float*)d_in[2];
  const float* in_proj_w  = (const float*)d_in[3];
  const float* in_proj_b  = (const float*)d_in[4];
  const float* out_proj_w = (const float*)d_in[5];
  const float* out_proj_b = (const float*)d_in[6];
  const float* ff1_w = (const float*)d_in[7];
  const float* ff1_b = (const float*)d_in[8];
  const float* ff2_w = (const float*)d_in[9];
  const float* ff2_b = (const float*)d_in[10];
  const float* ln1_w = (const float*)d_in[11];
  const float* ln1_b = (const float*)d_in[12];
  const float* ln2_w = (const float*)d_in[13];
  const float* ln2_b = (const float*)d_in[14];
  const float* hist_w = (const float*)d_in[15];
  const float* hist_b = (const float*)d_in[16];

  // Workspace carve-up (bytes):
  //   wbf16 (all weights)      : 786432*2 = 1,572,864
  //   te  f32 [4096*128]       : 2,097,152
  //   idx i32 [4096]           :    16,384
  //   xte f32 [4096*32*128]    : 67,108,864   -> total ~70.8 MB
  char* ws = (char*)d_ws;
  bf16* wall = (bf16*)(ws);
  bf16* wqkv = wall;                 // 2*768*256
  bf16* wo   = wall + 393216;        // 2*256*256
  bf16* wf1  = wall + 524288;
  bf16* wf2  = wall + 655360;
  float* te  = (float*)(ws + 1572864);
  int*   idx = (int*)(ws + 1572864 + 2097152);
  float* xte = (float*)(ws + 3686400);

  float* out_hist = (float*)d_out;
  float* out_npm  = out_hist + kHistElems;

  convert_kernel<<<786432 / 256, 256, 0, stream>>>(in_proj_w, out_proj_w, ff1_w, ff2_w, wall);
  prep_kernel<<<kInst, 128, 0, stream>>>(event_type, event_time, te, idx, out_npm);
  scan_kernel<<<kB * kT, 128, 0, stream>>>(te, idx, xte);
  tx_kernel<<<kInst, 128, 0, stream>>>(cat_emb, xte,
                                       wqkv, in_proj_b, wo, out_proj_b,
                                       wf1, ff1_b, wf2, ff2_b,
                                       ln1_w, ln1_b, ln2_w, ln2_b,
                                       hist_w, hist_b, out_hist);
}
